// LlamaAttention_20031727469032
// MI455X (gfx1250) — matrix operand order repaired and checked by reading, not compiled, not run on hardware
//
#include <hip/hip_runtime.h>
#include <hip/hip_bf16.h>

// ---------------------------------------------------------------------------
// LlamaAttention forward for gfx1250 (MI455X): bf16 WMMA, fp32 accumulate,
// fused RoPE, flash-attention with online softmax.
// 32x64 tiles per wave; ping-pong register buffering (2x unrolled K loop)
// so load/WMMA overlap costs zero register copies.
// B=2, S=2048, H=1024, NH=16, HD=64.
// ---------------------------------------------------------------------------

#define BB   2
#define SS   2048
#define HH   1024
#define NHH  16
#define HDD  64

typedef __attribute__((ext_vector_type(16))) __bf16 v16bf;
typedef __attribute__((ext_vector_type(8)))  __bf16 v8bf;
typedef __attribute__((ext_vector_type(8)))  float  v8f;

// D = A(16x32 bf16) * B(32x16 bf16) + C(16x16 f32)
__device__ __forceinline__ v8f wmma_bf16(v16bf a, v16bf b, v8f c) {
  return __builtin_amdgcn_wmma_f32_16x16x32_bf16(
      /*neg_a=*/false, a, /*neg_b=*/false, b,
      /*c_mod=*/(short)0, c, /*reuse_a=*/false, /*reuse_b=*/false);
}

// A fragment: 16x32 tile, row-major in memory, K contiguous.
// Lane L<16 : row L,    K = {0..7, 16..23}
// Lane L>=16: row L-16, K = {8..15, 24..31}
__device__ __forceinline__ v16bf load_a_frag(const __bf16* tile, int ld) {
  const int lane = threadIdx.x & 31;
  const int row  = lane & 15;
  const int koff = (lane >> 4) << 3;   // 0 or 8
  const __bf16* p = tile + row * ld + koff;
  v8bf lo = *(const v8bf*)(p);
  v8bf hi = *(const v8bf*)(p + 16);
  v16bf a;
#pragma unroll
  for (int i = 0; i < 8; ++i) { a[i] = lo[i]; a[i + 8] = hi[i]; }
  return a;
}

// B fragment: 32(K) x 16(N), with K contiguous per column in memory
// (i.e. source is row-major [N][K], an "NT" operand).
// Lane L<16 : col L,    K = {0..7, 16..23} ; Lane L>=16: col L-16, K = {8..15, 24..31}
__device__ __forceinline__ v16bf load_b_frag(const __bf16* tile, int ld) {
  const int lane = threadIdx.x & 31;
  const int col  = lane & 15;
  const int koff = (lane >> 4) << 3;   // 0 or 8
  const __bf16* p = tile + col * ld + koff;
  v8bf lo = *(const v8bf*)(p);
  v8bf hi = *(const v8bf*)(p + 16);
  v16bf b;
#pragma unroll
  for (int i = 0; i < 8; ++i) { b[i] = lo[i]; b[i + 8] = hi[i]; }
  return b;
}

// Row-wise reductions over the 16 N-lanes of a C/D register
// (xor masks < 16 never cross the half-wave boundary).
__device__ __forceinline__ float rowmax16(float x) {
  x = fmaxf(x, __shfl_xor(x, 1));
  x = fmaxf(x, __shfl_xor(x, 2));
  x = fmaxf(x, __shfl_xor(x, 4));
  x = fmaxf(x, __shfl_xor(x, 8));
  return x;
}
__device__ __forceinline__ float rowsum16(float x) {
  x += __shfl_xor(x, 1);
  x += __shfl_xor(x, 2);
  x += __shfl_xor(x, 4);
  x += __shfl_xor(x, 8);
  return x;
}

// ---------------------------------------------------------------------------
// Kernel 0: fp32 -> bf16 conversion (grid-stride)
// ---------------------------------------------------------------------------
__global__ void cvt_bf16_kernel(const float* __restrict__ src,
                                __bf16* __restrict__ dst, int n) {
  for (int i = blockIdx.x * blockDim.x + threadIdx.x; i < n;
       i += gridDim.x * blockDim.x)
    dst[i] = (__bf16)src[i];
}

// ---------------------------------------------------------------------------
// Kernel 1: QKV projection + fused RoPE (+ V stored transposed).
// grid = (B*S/32, NH, 3); one wave computes a 32(seq) x 64(head-dim) tile.
// K loop unrolled 2x with ping-pong fragment sets (no register copies).
// ---------------------------------------------------------------------------
__global__ __launch_bounds__(32)
void qkv_rope_kernel(const __bf16* __restrict__ hsb,
                     const __bf16* __restrict__ wq,
                     const __bf16* __restrict__ wk,
                     const __bf16* __restrict__ wv,
                     __bf16* __restrict__ Qb,
                     __bf16* __restrict__ Kb,
                     __bf16* __restrict__ Vtb) {
  const int r0   = blockIdx.x * 32;          // global row in [0, B*S)
  const int h    = blockIdx.y;
  const int z    = blockIdx.z;               // 0=Q 1=K 2=V
  const int b    = r0 >> 11;                 // / S
  const int s0   = r0 & (SS - 1);
  const int lane = threadIdx.x & 31;
  const int half = lane >> 4;
  const int c    = lane & 15;

  const __bf16* wb = (z == 0) ? wq : (z == 1) ? wk : wv;
  const __bf16* wh = wb + (size_t)h * HDD * HH;

  v8f acc[2][4];
#pragma unroll
  for (int mi = 0; mi < 2; ++mi)
#pragma unroll
    for (int j = 0; j < 4; ++j) acc[mi][j] = v8f{0};

  v16bf aA[2], bA[4], aB[2], bB[4];
#pragma unroll
  for (int mi = 0; mi < 2; ++mi)
    aA[mi] = load_a_frag(hsb + (size_t)(r0 + 16 * mi) * HH, HH);
#pragma unroll
  for (int j = 0; j < 4; ++j)
    bA[j] = load_b_frag(wh + (size_t)(j * 16) * HH, HH);

  for (int k = 0; k < HH; k += 64) {        // HH % 64 == 0
    // prefetch odd half-step (k+32) into set B  (always in range)
#pragma unroll
    for (int mi = 0; mi < 2; ++mi)
      aB[mi] = load_a_frag(hsb + (size_t)(r0 + 16 * mi) * HH + k + 32, HH);
#pragma unroll
    for (int j = 0; j < 4; ++j)
      bB[j] = load_b_frag(wh + (size_t)(j * 16) * HH + k + 32, HH);

#pragma unroll
    for (int j = 0; j < 4; ++j) {
      acc[0][j] = wmma_bf16(aA[0], bA[j], acc[0][j]);
      acc[1][j] = wmma_bf16(aA[1], bA[j], acc[1][j]);
    }

    // prefetch next even step (k+64) into set A
    if (k + 64 < HH) {
#pragma unroll
      for (int mi = 0; mi < 2; ++mi)
        aA[mi] = load_a_frag(hsb + (size_t)(r0 + 16 * mi) * HH + k + 64, HH);
#pragma unroll
      for (int j = 0; j < 4; ++j)
        bA[j] = load_b_frag(wh + (size_t)(j * 16) * HH + k + 64, HH);
    }

#pragma unroll
    for (int j = 0; j < 4; ++j) {
      acc[0][j] = wmma_bf16(aB[0], bB[j], acc[0][j]);
      acc[1][j] = wmma_bf16(aB[1], bB[j], acc[1][j]);
    }
  }

  // Fused RoPE for Q and K: rotate_half pairs d <-> d^32, which is
  // accumulator tile j <-> j^2 at the same (lane, vgpr). Pure elementwise.
  if (z < 2) {
#pragma unroll
    for (int mi = 0; mi < 2; ++mi) {
      float old[4][8];
#pragma unroll
      for (int j = 0; j < 4; ++j)
#pragma unroll
        for (int g = 0; g < 8; ++g) old[j][g] = acc[mi][j][g];

#pragma unroll
      for (int j = 0; j < 4; ++j) {
        const int d = j * 16 + c;
        const float inv = __powf(10000.0f, -(float)(d & 31) / 32.0f);
#pragma unroll
        for (int g = 0; g < 8; ++g) {
          const int srow = s0 + 16 * mi + g + 8 * half;
          float sn, cs;
          __sincosf((float)srow * inv, &sn, &cs);
          const float rot = (j < 2) ? -old[j + 2][g] : old[j - 2][g];
          acc[mi][j][g] = old[j][g] * cs + rot * sn;
        }
      }
    }
  }

  // Store. Q,K: [b][h][s][d] bf16.  V: transposed [b][h][d][s] bf16 so that
  // the P @ V B-fragments in the attention kernel are contiguous loads.
  const int bh = b * NHH + h;
#pragma unroll
  for (int mi = 0; mi < 2; ++mi) {
#pragma unroll
    for (int j = 0; j < 4; ++j) {
#pragma unroll
      for (int g = 0; g < 8; ++g) {
        const int m = 16 * mi + g + 8 * half;
        const int d = j * 16 + c;
        const int s = s0 + m;
        const __bf16 val = (__bf16)acc[mi][j][g];
        if (z == 0)      Qb [((size_t)bh * SS + s) * HDD + d] = val;
        else if (z == 1) Kb [((size_t)bh * SS + s) * HDD + d] = val;
        else             Vtb[((size_t)bh * HDD + d) * SS + s] = val;
      }
    }
  }
}

// ---------------------------------------------------------------------------
// Kernel 2: causal flash-attention. grid = (S/32, NH, B); one wave handles
// 32 query rows for one head; 32 keys per step. K fragments shared across
// both 16-row score tiles; V fragments across both P tiles. Step loop is
// pair-unrolled with ping-pong K buffers (zero-copy double buffering);
// V fragments prefetched before the softmax VALU section.
// ---------------------------------------------------------------------------
__global__ __launch_bounds__(32)
void attn_kernel(const __bf16* __restrict__ Qb,
                 const __bf16* __restrict__ Kb,
                 const __bf16* __restrict__ Vtb,
                 __bf16* __restrict__ ctx) {
  __shared__ __attribute__((aligned(32))) __bf16 Pl[32][40];  // padded rows

  const int q0   = blockIdx.x * 32;
  const int h    = blockIdx.y;
  const int b    = blockIdx.z;
  const int bh   = b * NHH + h;
  const int lane = threadIdx.x & 31;
  const int half = lane >> 4;
  const int c    = lane & 15;

  const __bf16* Qh  = Qb  + (size_t)bh * SS * HDD;
  const __bf16* Kh  = Kb  + (size_t)bh * SS * HDD;
  const __bf16* Vth = Vtb + (size_t)bh * HDD * SS;

  // Q fragments: [mi][kk]  (rows q0+16*mi, head-dim chunk 32*kk)
  v16bf aq[2][2];
#pragma unroll
  for (int mi = 0; mi < 2; ++mi)
#pragma unroll
    for (int kk = 0; kk < 2; ++kk)
      aq[mi][kk] = load_a_frag(Qh + (size_t)(q0 + 16 * mi) * HDD + 32 * kk, HDD);

  v8f o[2][4];
#pragma unroll
  for (int mi = 0; mi < 2; ++mi)
#pragma unroll
    for (int j = 0; j < 4; ++j) o[mi][j] = v8f{0};

  float m_run[2][8], l_run[2][8];
#pragma unroll
  for (int mi = 0; mi < 2; ++mi)
#pragma unroll
    for (int g = 0; g < 8; ++g) { m_run[mi][g] = -1.0e30f; l_run[mi][g] = 0.0f; }

  const int kend = q0 + 32;                  // causal upper bound

  auto loadK = [&](int kc, v16bf (&dst)[2][2]) {
#pragma unroll
    for (int t = 0; t < 2; ++t)
#pragma unroll
      for (int kk = 0; kk < 2; ++kk)
        dst[t][kk] = load_b_frag(Kh + (size_t)(kc + 16 * t) * HDD + 32 * kk, HDD);
  };

  // One 32-key step: scores from `cur`, prefetch kc+32 into `nxt`,
  // online softmax via LDS-staged P, O += P @ V.
  auto step = [&](int kc, v16bf (&cur)[2][2], v16bf (&nxt)[2][2]) {
    // ---- S = Q K^T, four 16x16 tiles sharing K fragments ----
    v8f sc[2][2];
#pragma unroll
    for (int mi = 0; mi < 2; ++mi)
#pragma unroll
      for (int t = 0; t < 2; ++t) {
        v8f s = v8f{0};
        s = wmma_bf16(aq[mi][0], cur[t][0], s);
        s = wmma_bf16(aq[mi][1], cur[t][1], s);
        sc[mi][t] = s;
      }

    // ---- prefetch V fragments (consumed after softmax) ----
    v16bf bv[4];
#pragma unroll
    for (int j = 0; j < 4; ++j)
      bv[j] = load_b_frag(Vth + (size_t)(j * 16) * SS + kc, SS);

    // ---- prefetch next step's K fragments into the other buffer ----
    if (kc + 32 < kend) loadK(kc + 32, nxt);

    __syncthreads();   // WAR on Pl from previous step

    // ---- causal mask + online softmax; stage P tiles in LDS ----
#pragma unroll
    for (int mi = 0; mi < 2; ++mi) {
#pragma unroll
      for (int g = 0; g < 8; ++g) {
        const int qrow = q0 + 16 * mi + g + 8 * half;
        float x0 = sc[mi][0][g] * 0.125f;    // 1/sqrt(64)
        float x1 = sc[mi][1][g] * 0.125f;
        if (kc + c      > qrow) x0 = -1.0e30f;
        if (kc + 16 + c > qrow) x1 = -1.0e30f;
        const float tmax = rowmax16(fmaxf(x0, x1));
        const float nm   = fmaxf(m_run[mi][g], tmax);
        const float corr = __expf(m_run[mi][g] - nm);
        const float p0   = __expf(x0 - nm);
        const float p1   = __expf(x1 - nm);
        l_run[mi][g] = l_run[mi][g] * corr + rowsum16(p0 + p1);
        m_run[mi][g] = nm;
#pragma unroll
        for (int j = 0; j < 4; ++j) o[mi][j][g] *= corr;
        Pl[16 * mi + g + 8 * half][c]      = (__bf16)p0;
        Pl[16 * mi + g + 8 * half][16 + c] = (__bf16)p1;
      }
    }

    __syncthreads();   // P visible (and compiler-ordered) before re-load

    // ---- O += P(16x32) @ V(32x64); V is [d][s] so K(=s) is contiguous ----
#pragma unroll
    for (int mi = 0; mi < 2; ++mi) {
      const v16bf ap = load_a_frag(&Pl[16 * mi][0], 40);
#pragma unroll
      for (int j = 0; j < 4; ++j)
        o[mi][j] = wmma_bf16(ap, bv[j], o[mi][j]);
    }
  };

  v16bf bkA[2][2], bkB[2][2];
  loadK(0, bkA);

  int kc = 0;
  for (; kc + 64 <= kend; kc += 64) {        // pair-unrolled ping-pong
    step(kc,      bkA, bkB);
    step(kc + 32, bkB, bkA);
  }
  if (kc < kend) step(kc, bkA, bkB);         // odd trailing step

  // ---- normalize and store context as [b][s][h*64+d] bf16 ----
#pragma unroll
  for (int mi = 0; mi < 2; ++mi) {
#pragma unroll
    for (int j = 0; j < 4; ++j) {
#pragma unroll
      for (int g = 0; g < 8; ++g) {
        const int s = q0 + 16 * mi + g + 8 * half;
        const int d = j * 16 + c;
        const float val = o[mi][j][g] / l_run[mi][g];
        ctx[((size_t)(b * SS + s)) * HH + h * HDD + d] = (__bf16)val;
      }
    }
  }
}

// ---------------------------------------------------------------------------
// Kernel 3: output projection  out[4096,1024] = ctx(bf16) @ Wo^T (fp32 out)
// grid = (B*S/32, H/64); one wave per 32 x 64 tile, ping-pong buffered.
// ---------------------------------------------------------------------------
__global__ __launch_bounds__(32)
void outproj_kernel(const __bf16* __restrict__ ctx,
                    const __bf16* __restrict__ wo,
                    float* __restrict__ out) {
  const int r0    = blockIdx.x * 32;
  const int nbase = blockIdx.y * 64;
  const int lane  = threadIdx.x & 31;
  const int half  = lane >> 4;
  const int c     = lane & 15;

  v8f acc[2][4];
#pragma unroll
  for (int mi = 0; mi < 2; ++mi)
#pragma unroll
    for (int j = 0; j < 4; ++j) acc[mi][j] = v8f{0};

  v16bf aA[2], bA[4], aB[2], bB[4];
#pragma unroll
  for (int mi = 0; mi < 2; ++mi)
    aA[mi] = load_a_frag(ctx + (size_t)(r0 + 16 * mi) * HH, HH);
#pragma unroll
  for (int j = 0; j < 4; ++j)
    bA[j] = load_b_frag(wo + (size_t)(nbase + j * 16) * HH, HH);

  for (int k = 0; k < HH; k += 64) {        // HH % 64 == 0
#pragma unroll
    for (int mi = 0; mi < 2; ++mi)
      aB[mi] = load_a_frag(ctx + (size_t)(r0 + 16 * mi) * HH + k + 32, HH);
#pragma unroll
    for (int j = 0; j < 4; ++j)
      bB[j] = load_b_frag(wo + (size_t)(nbase + j * 16) * HH + k + 32, HH);

#pragma unroll
    for (int j = 0; j < 4; ++j) {
      acc[0][j] = wmma_bf16(aA[0], bA[j], acc[0][j]);
      acc[1][j] = wmma_bf16(aA[1], bA[j], acc[1][j]);
    }

    if (k + 64 < HH) {
#pragma unroll
      for (int mi = 0; mi < 2; ++mi)
        aA[mi] = load_a_frag(ctx + (size_t)(r0 + 16 * mi) * HH + k + 64, HH);
#pragma unroll
      for (int j = 0; j < 4; ++j)
        bA[j] = load_b_frag(wo + (size_t)(nbase + j * 16) * HH + k + 64, HH);
    }

#pragma unroll
    for (int j = 0; j < 4; ++j) {
      acc[0][j] = wmma_bf16(aB[0], bB[j], acc[0][j]);
      acc[1][j] = wmma_bf16(aB[1], bB[j], acc[1][j]);
    }
  }

#pragma unroll
  for (int mi = 0; mi < 2; ++mi) {
#pragma unroll
    for (int j = 0; j < 4; ++j) {
#pragma unroll
      for (int g = 0; g < 8; ++g) {
        const int m = 16 * mi + g + 8 * half;
        const int n = nbase + j * 16 + c;
        out[(size_t)(r0 + m) * HH + n] = acc[mi][j][g];
      }
    }
  }
}

// ---------------------------------------------------------------------------
// Host-side launcher
// ---------------------------------------------------------------------------
extern "C" void kernel_launch(void* const* d_in, const int* in_sizes, int n_in,
                              void* d_out, int out_size, void* d_ws, size_t ws_size,
                              hipStream_t stream) {
  (void)in_sizes; (void)n_in; (void)out_size; (void)ws_size;

  const float* hs = (const float*)d_in[0];
  // d_in[1]=cos, d_in[2]=sin, d_in[3]=mask: recomputed analytically / implicit
  const float* Wq = (const float*)d_in[4];
  const float* Wk = (const float*)d_in[5];
  const float* Wv = (const float*)d_in[6];
  const float* Wo = (const float*)d_in[7];

  const size_t HS_E = (size_t)BB * SS * HH;   // 4,194,304
  const size_t W_E  = (size_t)HH * HH;        // 1,048,576

  __bf16* wsb = (__bf16*)d_ws;
  __bf16* hsb = wsb;
  __bf16* wqb = hsb + HS_E;
  __bf16* wkb = wqb + W_E;
  __bf16* wvb = wkb + W_E;
  __bf16* wob = wvb + W_E;
  __bf16* Qb  = wob + W_E;
  __bf16* Kb  = Qb  + HS_E;
  __bf16* Vtb = Kb  + HS_E;
  __bf16* ctx = Vtb + HS_E;
  // total ws usage: (5*HS_E + 4*W_E) * 2 bytes = ~48 MB

  cvt_bf16_kernel<<<dim3(1024), dim3(256), 0, stream>>>(hs, hsb, (int)HS_E);
  cvt_bf16_kernel<<<dim3(512),  dim3(256), 0, stream>>>(Wq, wqb, (int)W_E);
  cvt_bf16_kernel<<<dim3(512),  dim3(256), 0, stream>>>(Wk, wkb, (int)W_E);
  cvt_bf16_kernel<<<dim3(512),  dim3(256), 0, stream>>>(Wv, wvb, (int)W_E);
  cvt_bf16_kernel<<<dim3(512),  dim3(256), 0, stream>>>(Wo, wob, (int)W_E);

  qkv_rope_kernel<<<dim3(BB * SS / 32, NHH, 3), dim3(32), 0, stream>>>(
      hsb, wqb, wkb, wvb, Qb, Kb, Vtb);

  attn_kernel<<<dim3(SS / 32, NHH, BB), dim3(32), 0, stream>>>(
      Qb, Kb, Vtb, ctx);

  outproj_kernel<<<dim3(BB * SS / 32, HH / 64), dim3(32), 0, stream>>>(
      ctx, wob, (float*)d_out);
}
